// SelfAttention_11003706212570
// MI455X (gfx1250) — matrix operand order repaired and checked by reading, not compiled, not run on hardware
//
#include <hip/hip_runtime.h>
#include <hip/hip_bf16.h>
#include <math.h>

// ---------------------------------------------------------------------------
// MI455X (gfx1250) self-attention, all GEMMs on v_wmma_f32_16x16x32_bf16.
//   Q/K/V = X W*            (f32 -> bf16 staged in LDS, bf16 out)
//   S = (Q K^T)/sqrt(D1)    (f32 out; 64MB total -> lives in the 192MB L2)
//   P = softmax_row(S)      (in-place f32 -> bf16)
//   O = P V                 (f32 out)
// Compute-bound: ~120 GFLOP vs ~80MB mandatory HBM traffic.
// Inner loop: 8x ds_load_b128 -> 4x v_wmma per wave per K-step.
// Staging: vectorized (global b128/b64 -> ds_store_b64), fixed per-thread
// coordinates so address math is strength-reduced out of the K loop.
// ---------------------------------------------------------------------------

typedef __attribute__((ext_vector_type(16))) __bf16 v16bf;
typedef __attribute__((ext_vector_type(8)))  __bf16 v8bf;
typedef __attribute__((ext_vector_type(4)))  __bf16 v4bf;
typedef __attribute__((ext_vector_type(4)))  float  v4f;
typedef __attribute__((ext_vector_type(8)))  float  v8f;

#define BM 128
#define BN 64
#define BK 32
#define LDK 40          // padded K-stride in LDS (80B: 16B-aligned, odd*16)
#define NTHREADS 256    // 8 wave32s

__device__ __forceinline__ v4bf pack4(float a, float b, float c, float d) {
    v4bf r;
    r[0] = (__bf16)a; r[1] = (__bf16)b; r[2] = (__bf16)c; r[3] = (__bf16)d;
    return r;
}

template <typename T> __device__ __forceinline__ v4bf load4bf(const T* p);
template <> __device__ __forceinline__ v4bf load4bf<float>(const float* p) {
    const v4f f = *(const v4f*)p;                 // global_load_b128
    return pack4(f[0], f[1], f[2], f[3]);
}
template <> __device__ __forceinline__ v4bf load4bf<__bf16>(const __bf16* p) {
    return *(const v4bf*)p;                       // global_load_b64
}

// C[z] = scale * A[z] * op(B[z])
//  A: row-major M x K (lda); TA in {float, __bf16}, converted to bf16 in LDS
//  B: TRANSB ? row-major N x K (use B^T) : row-major K x N
//  C: row-major M x N (ldc); TC in {float, __bf16}
template <typename TA, typename TB, typename TC, bool TRANSB>
__global__ __launch_bounds__(NTHREADS)
void gemm_wmma_bf16(const TA* __restrict__ A, const TB* __restrict__ B,
                    TC* __restrict__ C,
                    int M, int N, int K, int lda, int ldb, int ldc,
                    float scale, long sA, long sB, long sC)
{
    __shared__ __align__(16) __bf16 As[BM][LDK];   // A tile, row-major
    __shared__ __align__(16) __bf16 Bt[BN][LDK];   // B tile, TRANSPOSED: Bt[n][k]

    const TA* Ab = A + (size_t)blockIdx.z * sA;
    const TB* Bb = B + (size_t)blockIdx.z * sB;
    TC*       Cb = C + (size_t)blockIdx.z * sC;

    const int tid   = threadIdx.x;
    const int lane  = tid & 31;
    const int wid   = tid >> 5;     // 0..7
    const int waveM = wid & 3;      // 4 wave rows  (32 M-rows each)
    const int waveN = wid >> 2;     // 2 wave cols  (32 N-cols each)
    const int h     = lane >> 4;    // WMMA lane half
    const int r     = lane & 15;

    const int mBase = blockIdx.x * BM;
    const int nBase = blockIdx.y * BN;

    // fixed per-thread staging coordinates (4-element vectors)
    const int arow = tid >> 3;            // 0..31 (A and transposed-B rows)
    const int acol = (tid & 7) << 2;      // 0,4,...,28
    const int bn   = tid & 63;            // non-trans B: n
    const int bk4  = (tid >> 6) << 2;     // non-trans B: k {0,4,8,12}

    const TA* pA = Ab + (size_t)(mBase + arow) * lda + acol;

    v8f acc[2][2] = {};                   // wave computes 32x32 = 4 WMMA tiles

    for (int kBase = 0; kBase < K; kBase += BK) {
        // ---- stage A tile (BM x BK): b128/b64 loads -> ds_store_b64 ----
        #pragma unroll
        for (int it = 0; it < (BM * BK / 4) / NTHREADS; ++it)   // 4
            *(v4bf*)&As[arow + it * 32][acol] =
                load4bf(pA + (size_t)(it * 32) * lda + kBase);

        // ---- stage B tile transposed -> Bt[n][k] ----
        if (TRANSB) {
            // B is N x K row-major: vector reads along k (coalesced)
            #pragma unroll
            for (int it = 0; it < (BN * BK / 4) / NTHREADS; ++it)  // 2
                *(v4bf*)&Bt[arow + it * 32][acol] =
                    load4bf(Bb + (size_t)(nBase + arow + it * 32) * ldb + acol + kBase);
        } else {
            // B is K x N row-major: 4 coalesced column reads -> one b64 store
            #pragma unroll
            for (int it = 0; it < 2; ++it) {
                const int k0 = bk4 + it * 16 + kBase;
                v4bf v;
                #pragma unroll
                for (int j = 0; j < 4; ++j)
                    v[j] = (__bf16)(float)Bb[(size_t)(k0 + j) * ldb + (nBase + bn)];
                *(v4bf*)&Bt[bn][bk4 + it * 16] = v;
            }
        }
        // gfx1250 prefetch of the next K-slab (global_prefetch_b8)
        if (kBase + BK < K)
            __builtin_prefetch((const void*)(pA + (size_t)lda + kBase + BK), 0, 3);
        __syncthreads();

        // ---- fragments as vector LDS loads (ds_load_b128 x2 each) ----
        // A lane(h,r): row, K in [8h,8h+8) and [16+8h,16+8h+8)
        v16bf afrag[2];
        #pragma unroll
        for (int mt = 0; mt < 2; ++mt) {
            const int row = waveM * 32 + mt * 16 + r;
            const v8bf lo = *(const v8bf*)&As[row][h * 8];
            const v8bf hi = *(const v8bf*)&As[row][16 + h * 8];
            afrag[mt] = __builtin_shufflevector(lo, hi,
                0, 1, 2, 3, 4, 5, 6, 7, 8, 9, 10, 11, 12, 13, 14, 15);
        }
        // B lane(h,r): column, K in [8h,8h+8) and [16+8h,16+8h+8)
        v16bf bfrag[2];
        #pragma unroll
        for (int nt = 0; nt < 2; ++nt) {
            const int col = waveN * 32 + nt * 16 + r;
            const v8bf lo = *(const v8bf*)&Bt[col][h * 8];
            const v8bf hi = *(const v8bf*)&Bt[col][16 + h * 8];
            bfrag[nt] = __builtin_shufflevector(lo, hi,
                0, 1, 2, 3, 4, 5, 6, 7, 8, 9, 10, 11, 12, 13, 14, 15);
        }

        #pragma unroll
        for (int mt = 0; mt < 2; ++mt)
            #pragma unroll
            for (int nt = 0; nt < 2; ++nt)
                acc[mt][nt] = __builtin_amdgcn_wmma_f32_16x16x32_bf16(
                    false, afrag[mt], false, bfrag[nt], (short)0,
                    acc[mt][nt], false, false);

        __syncthreads();
    }

    // ---- write back: C/D layout — VGPR j: row j + 8h, col r ----
    #pragma unroll
    for (int mt = 0; mt < 2; ++mt) {
        #pragma unroll
        for (int j = 0; j < 8; ++j) {
            const int row = mBase + waveM * 32 + mt * 16 + j + 8 * h;
            const int c0  = nBase + waveN * 32 + r;
            Cb[(size_t)row * ldc + c0]      = (TC)(acc[mt][0][j] * scale);
            Cb[(size_t)row * ldc + c0 + 16] = (TC)(acc[mt][1][j] * scale);
        }
    }
}

// Row softmax over 2048 f32 values, writing bf16 IN PLACE (row base reused).
// All reads complete (registers) before the post-reduction barrier; writes
// only touch the first half of the row's f32 storage.
__global__ __launch_bounds__(NTHREADS)
void softmax_rows_f32_to_bf16(float* __restrict__ S, int ncols)
{
    float* row = S + ((size_t)blockIdx.y * gridDim.x + blockIdx.x) * (size_t)ncols;
    const int tid = threadIdx.x;

    // thread handles 8 contiguous floats: two float4 vector loads
    v4f x0 = *(const v4f*)(row + tid * 8);
    v4f x1 = *(const v4f*)(row + tid * 8 + 4);

    float m = fmaxf(fmaxf(fmaxf(x0[0], x0[1]), fmaxf(x0[2], x0[3])),
                    fmaxf(fmaxf(x1[0], x1[1]), fmaxf(x1[2], x1[3])));

    __shared__ float red[NTHREADS];
    red[tid] = m;
    __syncthreads();
    for (int s = NTHREADS / 2; s > 0; s >>= 1) {
        if (tid < s) red[tid] = fmaxf(red[tid], red[tid + s]);
        __syncthreads();
    }
    const float rowmax = red[0];
    __syncthreads();

    float lsum = 0.f;
    #pragma unroll
    for (int j = 0; j < 4; ++j) { x0[j] = __expf(x0[j] - rowmax); lsum += x0[j]; }
    #pragma unroll
    for (int j = 0; j < 4; ++j) { x1[j] = __expf(x1[j] - rowmax); lsum += x1[j]; }

    red[tid] = lsum;
    __syncthreads();
    for (int s = NTHREADS / 2; s > 0; s >>= 1) {
        if (tid < s) red[tid] += red[tid + s];
        __syncthreads();
    }
    const float inv = 1.f / red[0];
    __syncthreads();

    __bf16* out = (__bf16*)row;   // bf16 row occupies first half of f32 row
    *(v4bf*)(out + tid * 8)     = pack4(x0[0] * inv, x0[1] * inv, x0[2] * inv, x0[3] * inv);
    *(v4bf*)(out + tid * 8 + 4) = pack4(x1[0] * inv, x1[1] * inv, x1[2] * inv, x1[3] * inv);
}

extern "C" void kernel_launch(void* const* d_in, const int* in_sizes, int n_in,
                              void* d_out, int out_size, void* d_ws, size_t ws_size,
                              hipStream_t stream)
{
    (void)in_sizes; (void)n_in; (void)out_size; (void)ws_size;
    const float* X  = (const float*)d_in[0];   // [4, 2048, 1024]
    const float* WQ = (const float*)d_in[1];   // [1024, 1024]
    const float* WK = (const float*)d_in[2];
    const float* WV = (const float*)d_in[3];
    float* OUT = (float*)d_out;                // [4, 2048, 1024] f32

    const int Bn = 4, N = 2048, D = 1024;
    const int M  = Bn * N;                     // 8192 flattened rows

    // ws layout: Qb/Kb/Vb bf16 (16MB each) + S f32 (64MB) = 112MB
    __bf16* Qb = (__bf16*)d_ws;
    __bf16* Kb = Qb + (size_t)M * D;
    __bf16* Vb = Kb + (size_t)M * D;
    float*  S  = (float*)(Vb + (size_t)M * D); // [4][2048][2048] f32

    dim3 blk(NTHREADS);

    // ---- Phase 1: QKV projections ----
    dim3 gqkv(M / BM, D / BN, 1);
    gemm_wmma_bf16<float, float, __bf16, false><<<gqkv, blk, 0, stream>>>(
        X, WQ, Qb, M, D, D, D, D, D, 1.0f, 0, 0, 0);
    gemm_wmma_bf16<float, float, __bf16, false><<<gqkv, blk, 0, stream>>>(
        X, WK, Kb, M, D, D, D, D, D, 1.0f, 0, 0, 0);
    gemm_wmma_bf16<float, float, __bf16, false><<<gqkv, blk, 0, stream>>>(
        X, WV, Vb, M, D, D, D, D, D, 1.0f, 0, 0, 0);

    // ---- Phase 2: S = Q K^T / sqrt(D1), batched over z ----
    const float scale = 1.0f / sqrtf((float)D);   // 0.03125
    dim3 gs(N / BM, N / BN, Bn);
    gemm_wmma_bf16<__bf16, __bf16, float, true><<<gs, blk, 0, stream>>>(
        Qb, Kb, S, N, N, D, D, D, N, scale,
        (long)N * D, (long)N * D, (long)N * N);

    // ---- Phase 3: row softmax, f32 -> bf16 in place ----
    softmax_rows_f32_to_bf16<<<dim3(N, Bn), blk, 0, stream>>>(S, N);

    // ---- Phase 4: O = P V (P bf16 packed at row starts of S, stride 2N) ----
    dim3 go(N / BM, D / BN, Bn);
    gemm_wmma_bf16<__bf16, __bf16, float, false><<<go, blk, 0, stream>>>(
        (const __bf16*)S, Vb, OUT, N, D, N,
        /*lda: f32 row = 2N bf16 slots*/ 2 * N, D, D, 1.0f,
        (long)N * (2 * N), (long)N * D, (long)N * D);
}
